// VRWKV_SpatialMix_V7_85066122264777
// MI455X (gfx1250) — matrix operand order repaired and checked by reading, not compiled, not run on hardware
//
#include <hip/hip_runtime.h>

#define BB   4
#define HIMG 24
#define WIMG 24
#define TT   (HIMG * WIMG)      // 576
#define HH   12
#define NHD  64
#define CC   (HH * NHD)         // 768
#define MTOT (BB * TT)          // 2304
#define ETOT (MTOT * CC)        // 1769472

typedef _Float16 v8h  __attribute__((ext_vector_type(8)));
typedef _Float16 v16h __attribute__((ext_vector_type(16)));
typedef float    v8f  __attribute__((ext_vector_type(8)));
typedef unsigned int u32x4 __attribute__((ext_vector_type(4)));
typedef int i32x8 __attribute__((ext_vector_type(8)));
typedef int i32x4 __attribute__((ext_vector_type(4)));

__device__ __forceinline__ float sigmoidf_(float x) { return 1.0f / (1.0f + __expf(-x)); }

// ---------------------------------------------------------------------------
// TDM: issue a 2D tensor_load_to_lds (6-arg builtin on this toolchain).
// D# per CDNA5 ISA. data_size=2B. LDS padding:
// pad_interval=16 DWORDs (one 32-elem row = 64B), pad_amount=4 DWORDs (16B)
// -> LDS row stride 80B = 40 halves. OOB rows (row >= rows_avail) read zero.
// ---------------------------------------------------------------------------
__device__ __forceinline__ void tdm_load_tile_2d(unsigned lds_byte_off,
                                                 const _Float16* gsrc,
                                                 int rows_avail,
                                                 int row_stride_elems,
                                                 int tile_w, int tile_h) {
    unsigned long long ga = (unsigned long long)gsrc;
    u32x4 g0;
    g0.x = 1u;                                   // count=1 (valid), no gather
    g0.y = lds_byte_off;                         // lds_addr [63:32]
    g0.z = (unsigned)ga;                         // global_addr [95:64]
    g0.w = (unsigned)((ga >> 32) & 0x01FFFFFFu) | (2u << 30);  // addr[56:32] | type=2
    unsigned td0 = (unsigned)tile_w;             // tensor_dim0 == tile width (never OOB in K)
    unsigned td1 = (unsigned)(rows_avail < 0 ? 0 : rows_avail);
    i32x8 g1;
    g1[0] = (int)((1u << 16) | (1u << 20) | (3u << 22) | (3u << 25));
            // data_size=1(2B) | pad_enable | pad_interval=16DW | pad_amount=4DW
    g1[1] = (int)((td0 & 0xFFFFu) << 16);                       // tensor_dim0 [79:48] lo
    g1[2] = (int)((td0 >> 16) | ((td1 & 0xFFFFu) << 16));       // tensor_dim0 hi | dim1 lo
    g1[3] = (int)((td1 >> 16) | ((unsigned)tile_w << 16));      // dim1 hi | tile_dim0
    g1[4] = (int)((unsigned)tile_h & 0xFFFFu);                  // tile_dim1 (tile_dim2=0)
    g1[5] = (int)(unsigned)row_stride_elems;                    // tensor_dim0_stride lo32
    g1[6] = 0;
    g1[7] = 0;
    i32x4 gz4 = {0, 0, 0, 0};                   // 2D tensor: groups 2/3 unused
    i32x8 gz8 = {0, 0, 0, 0, 0, 0, 0, 0};       // unused trailing group
    __builtin_amdgcn_tensor_load_to_lds(g0, g1, gz4, gz4, gz8, 0);
}

// ---------------------------------------------------------------------------
// WMMA GEMM:  C(M,N) f32 = A(M,K) f16 row-major  @  B(N,K) f16 row-major ^T
// Block: 256 threads = 8 waves; macro-tile 128(M) x 64(N); each wave 32x32.
// K-step 32 staged by the Tensor Data Mover into double-buffered LDS
// (A: 128x32, B: 64x32, rows padded to 40 halves by TDM pad fields).
// Each wave DMAs its own slice (A: 16 rows, B: 8 rows) and waits its own
// TENSORcnt -> one workgroup barrier per K-step, DMA overlapped with WMMA.
// OOB B rows (N < 64 case) are zero-filled by TDM so WMMA runs with EXEC
// all-ones, unconditionally.
// Requires M % 128 == 0, K % 32 == 0, N % 16 == 0 (all satisfied here).
// ---------------------------------------------------------------------------
#define APAD 40
#define BPAD 40

__global__ __launch_bounds__(256)
void wmma_gemm_nt(const _Float16* __restrict__ A, const _Float16* __restrict__ Bm,
                  float* __restrict__ Cm, int M, int N, int K) {
    __shared__ _Float16 sA[2][128 * APAD];
    __shared__ _Float16 sB[2][64 * BPAD];

    const int tid  = threadIdx.x;
    const int lane = tid & 31;
    const int wave = __builtin_amdgcn_readfirstlane(tid >> 5);   // SGPR-uniform
    const int mblk = blockIdx.y * 128;
    const int nblk = blockIdx.x * 64;
    const int m0   = (wave >> 1) * 32;    // wave tile origin within block (M)
    const int n0   = (wave & 1)  * 32;    // wave tile origin within block (N)

    // per-wave TDM slices
    const int aslice_row = wave * 16;     // 16 A rows per wave
    const int bslice_row = wave * 8;      // 8 B rows per wave
    const _Float16* Aslice = A + (size_t)(mblk + aslice_row) * K;
    const _Float16* Bslice = Bm + (size_t)(nblk + bslice_row) * K;
    const int brows_avail  = N - (nblk + bslice_row);   // may be <= 0 -> zeros
    const unsigned sA_off0 = (unsigned)(unsigned long long)&sA[0][0];
    const unsigned sB_off0 = (unsigned)(unsigned long long)&sB[0][0];
    const unsigned aoff_w  = (unsigned)(aslice_row * APAD * 2);
    const unsigned boff_w  = (unsigned)(bslice_row * BPAD * 2);
    const unsigned bufA_sz = (unsigned)(128 * APAD * 2);
    const unsigned bufB_sz = (unsigned)(64 * BPAD * 2);

    // fragment addressing (ISA 7.12.2):
    // A 16x32: lanes 0-15 row m, K={0..7,16..23}; lanes 16-31 row m, K={8..15,24..31}
    // B 32x16: lanes 0-15 col n, K={0..7,16..23}; lanes 16-31 col n, K={8..15,24..31}
    const int frow  = lane & 15;
    const int akoff = (lane >> 4) ? 8 : 0;
    const int bkoff = (lane >> 4) ? 8 : 0;

    v8f acc00 = {}, acc01 = {}, acc10 = {}, acc11 = {};

    const int nIter = K >> 5;    // K/32 steps
    // prologue: DMA first K-step into buffer 0
    tdm_load_tile_2d(sA_off0 + aoff_w, Aslice, 16, K, 32, 16);
    tdm_load_tile_2d(sB_off0 + boff_w, Bslice, brows_avail, K, 32, 8);

    for (int it = 0; it < nIter; ++it) {
        const int cur = it & 1;
        __builtin_amdgcn_s_wait_tensorcnt(0);   // this wave's slice landed
        __syncthreads();                        // all waves' slices landed
        if (it + 1 < nIter) {                   // DMA next step into other buffer
            const int k1 = (it + 1) << 5;
            const int nxt = cur ^ 1;
            tdm_load_tile_2d(sA_off0 + (unsigned)nxt * bufA_sz + aoff_w, Aslice + k1, 16, K, 32, 16);
            tdm_load_tile_2d(sB_off0 + (unsigned)nxt * bufB_sz + boff_w, Bslice + k1, brows_avail, K, 32, 8);
        }

        v16h a0, a1, b0, b1;
        {
            const _Float16* pa0 = &sA[cur][(m0 + frow) * APAD + akoff];
            const _Float16* pa1 = &sA[cur][(m0 + 16 + frow) * APAD + akoff];
            v8h lo0 = *(const v8h*)pa0, hi0 = *(const v8h*)(pa0 + 16);
            v8h lo1 = *(const v8h*)pa1, hi1 = *(const v8h*)(pa1 + 16);
            const _Float16* pb0 = &sB[cur][(n0 + frow) * BPAD + bkoff];
            const _Float16* pb1 = &sB[cur][(n0 + 16 + frow) * BPAD + bkoff];
            v8h b0l = *(const v8h*)pb0, b0h = *(const v8h*)(pb0 + 16);
            v8h b1l = *(const v8h*)pb1, b1h = *(const v8h*)(pb1 + 16);
#pragma unroll
            for (int i = 0; i < 8; ++i) {
                a0[i] = lo0[i]; a0[8 + i] = hi0[i];
                a1[i] = lo1[i]; a1[8 + i] = hi1[i];
                b0[i] = b0l[i]; b0[8 + i] = b0h[i];
                b1[i] = b1l[i]; b1[8 + i] = b1h[i];
            }
        }
        acc00 = __builtin_amdgcn_wmma_f32_16x16x32_f16(false, a0, false, b0, (short)0, acc00, false, false);
        acc10 = __builtin_amdgcn_wmma_f32_16x16x32_f16(false, a1, false, b0, (short)0, acc10, false, false);
        acc01 = __builtin_amdgcn_wmma_f32_16x16x32_f16(false, a0, false, b1, (short)0, acc01, false, false);
        acc11 = __builtin_amdgcn_wmma_f32_16x16x32_f16(false, a1, false, b1, (short)0, acc11, false, false);
    }

    // C/D layout: VGPR g, lanes 0-15 -> (M=g, N=lane); lanes 16-31 -> (M=8+g, N=lane-16)
    const int crow = (lane >> 4) ? 8 : 0;
    const int ccol = lane & 15;
    const int gm0  = mblk + m0;
    const int gn0  = nblk + n0;
    if (gn0 < N) {
#pragma unroll
        for (int g = 0; g < 8; ++g) {
            Cm[(size_t)(gm0 + crow + g) * N + gn0 + ccol]      = acc00[g];
            Cm[(size_t)(gm0 + 16 + crow + g) * N + gn0 + ccol] = acc10[g];
        }
    }
    if (gn0 + 16 < N) {
#pragma unroll
        for (int g = 0; g < 8; ++g) {
            Cm[(size_t)(gm0 + crow + g) * N + gn0 + 16 + ccol]      = acc01[g];
            Cm[(size_t)(gm0 + 16 + crow + g) * N + gn0 + 16 + ccol] = acc11[g];
        }
    }
}

// ---------------------------------------------------------------------------
// Conversions
// ---------------------------------------------------------------------------
__global__ void cvt_f16(const float* __restrict__ in, _Float16* __restrict__ out, int n) {
    int i = blockIdx.x * blockDim.x + threadIdx.x;
    if (i < n) out[i] = (_Float16)in[i];
}

// in: (K,N) row-major f32  ->  out: (N,K) row-major f16
__global__ void cvt_f16_T(const float* __restrict__ in, _Float16* __restrict__ out, int K, int N) {
    int i = blockIdx.x * blockDim.x + threadIdx.x;
    if (i >= K * N) return;
    int n = i / K, k = i % K;
    out[i] = (_Float16)in[(size_t)k * N + n];
}

__global__ void tanh_ip(float* p, int n) {
    int i = blockIdx.x * blockDim.x + threadIdx.x;
    if (i < n) p[i] = tanhf(p[i]);
}
__global__ void sigmoid_ip(float* p, int n) {
    int i = blockIdx.x * blockDim.x + threadIdx.x;
    if (i < n) p[i] = sigmoidf_(p[i]);
}

// ---------------------------------------------------------------------------
// q_shift: xx = qshift(x) - x ;  xxx16 = f16(x + xx*maa_x)
// ---------------------------------------------------------------------------
__global__ void qshift_mix(const float* __restrict__ x, const float* __restrict__ maa_x,
                           float* __restrict__ xx, _Float16* __restrict__ xxx16) {
    int idx = blockIdx.x * blockDim.x + threadIdx.x;
    if (idx >= ETOT) return;
    int c = idx % CC;
    int t = (idx / CC) % TT;
    int b = idx / (CC * TT);
    int hi = t / WIMG, wi = t % WIMG;
    const int q = CC / 4;
    float sv = 0.0f;
    if (c < q)            { if (wi > 0)        sv = x[((size_t)(b * TT + hi * WIMG + wi - 1)) * CC + c]; }
    else if (c < 2 * q)   { if (wi < WIMG - 1) sv = x[((size_t)(b * TT + hi * WIMG + wi + 1)) * CC + c]; }
    else if (c < 3 * q)   { if (hi > 0)        sv = x[((size_t)(b * TT + (hi - 1) * WIMG + wi)) * CC + c]; }
    else                  { if (hi < HIMG - 1) sv = x[((size_t)(b * TT + (hi + 1) * WIMG + wi)) * CC + c]; }
    float xv = x[idx];
    float d  = sv - xv;
    xx[idx]  = d;
    xxx16[idx] = (_Float16)(xv + d * maa_x[c]);
}

// ---------------------------------------------------------------------------
// Fused LoRA second stages (K small -> fp32 dot per output element)
// ---------------------------------------------------------------------------
__global__ void stage2_mix(const float* __restrict__ x, const float* __restrict__ xx,
                           const float* __restrict__ lora, int foff,
                           const float* __restrict__ w2, const float* __restrict__ maa,
                           _Float16* __restrict__ out16) {
    int idx = blockIdx.x * blockDim.x + threadIdx.x;
    if (idx >= ETOT) return;
    int m = idx / CC, c = idx % CC;
    const float* lp = lora + (size_t)m * 128 + foff;
    float acc = maa[c];
#pragma unroll 8
    for (int d = 0; d < 32; ++d) acc += lp[d] * w2[(size_t)d * CC + c];
    out16[idx] = (_Float16)(x[idx] + xx[idx] * acc);
}

// wlog = -softplus(-(w0 + wtan @ wd2)) - 0.5
__global__ void stage2_wdecay(const float* __restrict__ wtan, const float* __restrict__ wd2,
                              const float* __restrict__ w0, float* __restrict__ wlog) {
    int idx = blockIdx.x * blockDim.x + threadIdx.x;
    if (idx >= ETOT) return;
    int m = idx / CC, c = idx % CC;
    const float* wp = wtan + (size_t)m * 64;
    float z = w0[c];
#pragma unroll 8
    for (int d = 0; d < 64; ++d) z += wp[d] * wd2[(size_t)d * CC + c];
    float nz = -z;
    float sp = (nz > 20.0f) ? nz : __logf(1.0f + __expf(nz));
    wlog[idx] = -sp - 0.5f;
}

// g = gsig @ g2
__global__ void stage2_g(const float* __restrict__ gsig, const float* __restrict__ g2,
                         float* __restrict__ g) {
    int idx = blockIdx.x * blockDim.x + threadIdx.x;
    if (idx >= ETOT) return;
    int m = idx / CC, c = idx % CC;
    const float* gp = gsig + (size_t)m * 128;
    float acc = 0.0f;
#pragma unroll 8
    for (int d = 0; d < 128; ++d) acc += gp[d] * g2[(size_t)d * CC + c];
    g[idx] = acc;
}

// kk = kraw + kktan @ kk2
__global__ void stage2_kk(const float* __restrict__ kraw, const float* __restrict__ kktan,
                          const float* __restrict__ kk2, float* __restrict__ kk) {
    int idx = blockIdx.x * blockDim.x + threadIdx.x;
    if (idx >= ETOT) return;
    int m = idx / CC, c = idx % CC;
    const float* kp = kktan + (size_t)m * 16;
    float acc = kraw[idx];
#pragma unroll
    for (int d = 0; d < 16; ++d) acc += kp[d] * kk2[(size_t)d * CC + c];
    kk[idx] = acc;
}

// a = sigmoid(a0 + apre @ a2)
__global__ void stage2_asig(const float* __restrict__ apre, const float* __restrict__ a2,
                            const float* __restrict__ a0, float* __restrict__ a) {
    int idx = blockIdx.x * blockDim.x + threadIdx.x;
    if (idx >= ETOT) return;
    int m = idx / CC, c = idx % CC;
    const float* ap = apre + (size_t)m * 16;
    float acc = a0[c];
#pragma unroll
    for (int d = 0; d < 16; ++d) acc += ap[d] * a2[(size_t)d * CC + c];
    a[idx] = sigmoidf_(acc);
}

// k = (k*ma + k*a*(1-ma)) * exp(min(wlog*mk, 0));  ma/mk LoRA fused in
__global__ void kfinal(float* __restrict__ k, const float* __restrict__ a,
                       const float* __restrict__ wlog,
                       const float* __restrict__ mapre, const float* __restrict__ mkpre,
                       const float* __restrict__ ma2, const float* __restrict__ mk2,
                       const float* __restrict__ ma0, const float* __restrict__ mk0) {
    int idx = blockIdx.x * blockDim.x + threadIdx.x;
    if (idx >= ETOT) return;
    int m = idx / CC, c = idx % CC;
    const float* map = mapre + (size_t)m * 16;
    const float* mkp = mkpre + (size_t)m * 16;
    float za = ma0[c], zk = mk0[c];
#pragma unroll
    for (int d = 0; d < 16; ++d) {
        za += map[d] * ma2[(size_t)d * CC + c];
        zk += mkp[d] * mk2[(size_t)d * CC + c];
    }
    float ma = sigmoidf_(za);
    float mk = sigmoidf_(zk);
    float kv = k[idx];
    float av = a[idx];
    kv = kv * ma + kv * av * (1.0f - ma);
    float e = wlog[idx] * mk;
    if (e > 0.0f) e = 0.0f;
    k[idx] = kv * __expf(e);
}

// ---------------------------------------------------------------------------
// 64-thread block reduction helper
// ---------------------------------------------------------------------------
__device__ __forceinline__ float blockReduce64(float v, float* buf) {
    int t = threadIdx.x;
    buf[t] = v;
    __syncthreads();
#pragma unroll
    for (int s = 32; s > 0; s >>= 1) {
        if (t < s) buf[t] += buf[t + s];
        __syncthreads();
    }
    float r = buf[0];
    __syncthreads();
    return r;
}

// kk head-wise L2 normalize, in place.  grid: MTOT*HH blocks x 64 threads
__global__ __launch_bounds__(64) void kknorm(float* __restrict__ kk) {
    int mh = blockIdx.x;
    int m = mh / HH, h = mh % HH, n = threadIdx.x;
    size_t idx = (size_t)m * CC + h * NHD + n;
    __shared__ float red[NHD];
    float v = kk[idx];
    float ss = blockReduce64(v * v, red);
    float denom = fmaxf(sqrtf(ss), 1e-12f);
    kk[idx] = v / denom;
}

// ---------------------------------------------------------------------------
// RWKV-7 scan: grid BB*HH blocks x 64 threads; thread i owns state row i (64 regs)
// S_ij = S_ij*d_j + (S a)_i * b_j + v_i * k_j ;  y_i = sum_j S_ij r_j
// a = -kk, b = kk*asig
// ---------------------------------------------------------------------------
__global__ __launch_bounds__(64)
void wkv7_scan(const float* __restrict__ r, const float* __restrict__ wlog,
               const float* __restrict__ k, const float* __restrict__ v,
               const float* __restrict__ kk, const float* __restrict__ asig,
               float* __restrict__ y) {
    int b = blockIdx.x / HH;
    int h = blockIdx.x % HH;
    int i = threadIdx.x;
    float S[NHD];
#pragma unroll
    for (int j = 0; j < NHD; ++j) S[j] = 0.0f;

    __shared__ float sR[NHD], sD[NHD], sK[NHD], sA[NHD], sB2[NHD];

    for (int t = 0; t < TT; ++t) {
        size_t base = ((size_t)(b * TT + t)) * CC + h * NHD;
        float rv  = r[base + i];
        float dv  = __expf(wlog[base + i]);
        float kv  = k[base + i];
        float kkv = kk[base + i];
        float av  = asig[base + i];
        float vi  = v[base + i];
        sR[i]  = rv;
        sD[i]  = dv;
        sK[i]  = kv;
        sA[i]  = -kkv;       // a vector
        sB2[i] = kkv * av;   // b vector
        __syncthreads();

        float sa = 0.0f;
#pragma unroll
        for (int j = 0; j < NHD; ++j) sa += S[j] * sA[j];

        float yv = 0.0f;
#pragma unroll
        for (int j = 0; j < NHD; ++j) {
            float s = S[j] * sD[j] + sa * sB2[j] + vi * sK[j];
            S[j] = s;
            yv += s * sR[j];
        }
        y[base + i] = yv;
        __syncthreads();
    }
}

// ---------------------------------------------------------------------------
// GroupNorm + bonus + gate:  yg16 = f16( (GN(y)*s+b + (sum_n r*k*faaaa)*v) * g )
// grid: MTOT*HH blocks x 64 threads
// ---------------------------------------------------------------------------
__global__ __launch_bounds__(64)
void gn_bonus_gate(const float* __restrict__ y, const float* __restrict__ r,
                   const float* __restrict__ k, const float* __restrict__ v,
                   const float* __restrict__ g, const float* __restrict__ lns,
                   const float* __restrict__ lnb, const float* __restrict__ faaaa,
                   _Float16* __restrict__ yg16) {
    int mh = blockIdx.x;
    int m = mh / HH, h = mh % HH, n = threadIdx.x;
    int c = h * NHD + n;
    size_t idx = (size_t)m * CC + c;
    __shared__ float red[NHD];
    float yv = y[idx];
    float mu  = blockReduce64(yv, red) * (1.0f / NHD);
    float var = blockReduce64(yv * yv, red) * (1.0f / NHD) - mu * mu;
    float xn  = (yv - mu) * rsqrtf(var + 64e-5f);
    float yn  = xn * lns[c] + lnb[c];
    float rkf = r[idx] * k[idx] * faaaa[c];
    float coef = blockReduce64(rkf, red);
    yn += coef * v[idx];
    yg16[idx] = (_Float16)(yn * g[idx]);
}

// ---------------------------------------------------------------------------
// Host launcher
// ---------------------------------------------------------------------------
extern "C" void kernel_launch(void* const* d_in, const int* in_sizes, int n_in,
                              void* d_out, int out_size, void* d_ws, size_t ws_size,
                              hipStream_t stream) {
    (void)in_sizes; (void)n_in; (void)out_size; (void)ws_size;
    const float* x      = (const float*)d_in[0];
    const float* maa_x  = (const float*)d_in[1];
    const float* maa_rg = (const float*)d_in[2];
    const float* maa_wa = (const float*)d_in[3];
    const float* maa_k  = (const float*)d_in[4];
    const float* maa_v  = (const float*)d_in[5];
    const float* maa_w1 = (const float*)d_in[6];
    const float* maa_w2 = (const float*)d_in[7];
    const float* w0     = (const float*)d_in[8];
    const float* wd1    = (const float*)d_in[9];
    const float* wd2    = (const float*)d_in[10];
    const float* a0     = (const float*)d_in[11];
    const float* a1     = (const float*)d_in[12];
    const float* a2     = (const float*)d_in[13];
    const float* kk1    = (const float*)d_in[14];
    const float* kk2    = (const float*)d_in[15];
    const float* g1     = (const float*)d_in[16];
    const float* g2     = (const float*)d_in[17];
    const float* ma0    = (const float*)d_in[18];
    const float* ma1    = (const float*)d_in[19];
    const float* ma2    = (const float*)d_in[20];
    const float* mk0    = (const float*)d_in[21];
    const float* mk1    = (const float*)d_in[22];
    const float* mk2    = (const float*)d_in[23];
    const float* faaaa  = (const float*)d_in[24];
    const float* Wr     = (const float*)d_in[25];
    const float* Wk     = (const float*)d_in[26];
    const float* Wv     = (const float*)d_in[27];
    const float* Wo     = (const float*)d_in[28];
    const float* ln_s   = (const float*)d_in[29];
    const float* ln_b   = (const float*)d_in[30];
    float* out = (float*)d_out;

    char* wsb = (char*)d_ws;
    size_t off = 0;
    auto carve = [&](size_t bytes) -> char* {
        char* p = wsb + off;
        off += (bytes + 255) & ~(size_t)255;
        return p;
    };

    // f32 activations
    float* xx    = (float*)carve((size_t)ETOT * 4);
    float* rbuf  = (float*)carve((size_t)ETOT * 4);
    float* kbuf  = (float*)carve((size_t)ETOT * 4);  // kraw -> k (in place)
    float* vbuf  = (float*)carve((size_t)ETOT * 4);
    float* wlog  = (float*)carve((size_t)ETOT * 4);
    float* gbuf  = (float*)carve((size_t)ETOT * 4);
    float* kkbuf = (float*)carve((size_t)ETOT * 4);
    float* abuf  = (float*)carve((size_t)ETOT * 4);
    float* ybuf  = (float*)carve((size_t)ETOT * 4);
    // f16 activations
    _Float16* xxx16 = (_Float16*)carve((size_t)ETOT * 2);
    _Float16* xrg16 = (_Float16*)carve((size_t)ETOT * 2);
    _Float16* xwa16 = (_Float16*)carve((size_t)ETOT * 2);
    _Float16* xk16  = (_Float16*)carve((size_t)ETOT * 2);
    _Float16* xv16  = (_Float16*)carve((size_t)ETOT * 2);
    _Float16* yg16  = (_Float16*)carve((size_t)ETOT * 2);
    // small stage-1 outputs (f32)
    float* lora  = (float*)carve((size_t)MTOT * 128 * 4);
    float* wtan  = (float*)carve((size_t)MTOT * 64 * 4);
    float* gsig  = (float*)carve((size_t)MTOT * 128 * 4);
    float* kktan = (float*)carve((size_t)MTOT * 16 * 4);
    float* apre  = (float*)carve((size_t)MTOT * 16 * 4);
    float* mapre = (float*)carve((size_t)MTOT * 16 * 4);
    float* mkpre = (float*)carve((size_t)MTOT * 16 * 4);
    // f16 weights
    _Float16* Wr16  = (_Float16*)carve((size_t)CC * CC * 2);
    _Float16* Wk16  = (_Float16*)carve((size_t)CC * CC * 2);
    _Float16* Wv16  = (_Float16*)carve((size_t)CC * CC * 2);
    _Float16* Wo16  = (_Float16*)carve((size_t)CC * CC * 2);
    _Float16* w1T   = (_Float16*)carve((size_t)128 * CC * 2);
    _Float16* wd1T  = (_Float16*)carve((size_t)64 * CC * 2);
    _Float16* g1T   = (_Float16*)carve((size_t)128 * CC * 2);
    _Float16* a1T   = (_Float16*)carve((size_t)16 * CC * 2);
    _Float16* kk1T  = (_Float16*)carve((size_t)16 * CC * 2);
    _Float16* ma1T  = (_Float16*)carve((size_t)16 * CC * 2);
    _Float16* mk1T  = (_Float16*)carve((size_t)16 * CC * 2);

    const int TPB = 256;
    const int EBLK = (ETOT + TPB - 1) / TPB;
    const int WBLK = (CC * CC + TPB - 1) / TPB;

    auto gemm = [&](const _Float16* A, const _Float16* Bm, float* Cm, int M, int N, int K) {
        dim3 grid((N + 63) / 64, M / 128);
        wmma_gemm_nt<<<grid, dim3(256), 0, stream>>>(A, Bm, Cm, M, N, K);
    };

    // --- weight conversion ---
    cvt_f16<<<WBLK, TPB, 0, stream>>>(Wr, Wr16, CC * CC);
    cvt_f16<<<WBLK, TPB, 0, stream>>>(Wk, Wk16, CC * CC);
    cvt_f16<<<WBLK, TPB, 0, stream>>>(Wv, Wv16, CC * CC);
    cvt_f16<<<WBLK, TPB, 0, stream>>>(Wo, Wo16, CC * CC);
    cvt_f16_T<<<(128 * CC + TPB - 1) / TPB, TPB, 0, stream>>>(maa_w1, w1T, CC, 128);
    cvt_f16_T<<<(64 * CC + TPB - 1) / TPB, TPB, 0, stream>>>(wd1, wd1T, CC, 64);
    cvt_f16_T<<<(128 * CC + TPB - 1) / TPB, TPB, 0, stream>>>(g1, g1T, CC, 128);
    cvt_f16_T<<<(16 * CC + TPB - 1) / TPB, TPB, 0, stream>>>(a1, a1T, CC, 16);
    cvt_f16_T<<<(16 * CC + TPB - 1) / TPB, TPB, 0, stream>>>(kk1, kk1T, CC, 16);
    cvt_f16_T<<<(16 * CC + TPB - 1) / TPB, TPB, 0, stream>>>(ma1, ma1T, CC, 16);
    cvt_f16_T<<<(16 * CC + TPB - 1) / TPB, TPB, 0, stream>>>(mk1, mk1T, CC, 16);

    // --- token shift & mix LoRA ---
    qshift_mix<<<EBLK, TPB, 0, stream>>>(x, maa_x, xx, xxx16);
    gemm(xxx16, w1T, lora, MTOT, 128, CC);
    tanh_ip<<<(MTOT * 128 + TPB - 1) / TPB, TPB, 0, stream>>>(lora, MTOT * 128);
    stage2_mix<<<EBLK, TPB, 0, stream>>>(x, xx, lora, 0,  maa_w2 + 0 * 32 * CC, maa_rg, xrg16);
    stage2_mix<<<EBLK, TPB, 0, stream>>>(x, xx, lora, 32, maa_w2 + 1 * 32 * CC, maa_wa, xwa16);
    stage2_mix<<<EBLK, TPB, 0, stream>>>(x, xx, lora, 64, maa_w2 + 2 * 32 * CC, maa_k, xk16);
    stage2_mix<<<EBLK, TPB, 0, stream>>>(x, xx, lora, 96, maa_w2 + 3 * 32 * CC, maa_v, xv16);

    // --- big projections (WMMA) ---
    gemm(xrg16, Wr16, rbuf, MTOT, CC, CC);
    gemm(xk16,  Wk16, kbuf, MTOT, CC, CC);
    gemm(xv16,  Wv16, vbuf, MTOT, CC, CC);

    // --- decay w ---
    gemm(xwa16, wd1T, wtan, MTOT, 64, CC);
    tanh_ip<<<(MTOT * 64 + TPB - 1) / TPB, TPB, 0, stream>>>(wtan, MTOT * 64);
    stage2_wdecay<<<EBLK, TPB, 0, stream>>>(wtan, wd2, w0, wlog);

    // --- gate g ---
    gemm(xrg16, g1T, gsig, MTOT, 128, CC);
    sigmoid_ip<<<(MTOT * 128 + TPB - 1) / TPB, TPB, 0, stream>>>(gsig, MTOT * 128);
    stage2_g<<<EBLK, TPB, 0, stream>>>(gsig, g2, gbuf);

    // --- kk (normalized removal key) ---
    gemm(xk16, kk1T, kktan, MTOT, 16, CC);
    tanh_ip<<<(MTOT * 16 + TPB - 1) / TPB, TPB, 0, stream>>>(kktan, MTOT * 16);
    stage2_kk<<<EBLK, TPB, 0, stream>>>(kbuf, kktan, kk2, kkbuf);
    kknorm<<<MTOT * HH, 64, 0, stream>>>(kkbuf);

    // --- a (in-context learning rate) ---
    gemm(xwa16, a1T, apre, MTOT, 16, CC);
    stage2_asig<<<EBLK, TPB, 0, stream>>>(apre, a2, a0, abuf);

    // --- ma / mk LoRA pre-products, final k ---
    gemm(xwa16, ma1T, mapre, MTOT, 16, CC);
    gemm(xwa16, mk1T, mkpre, MTOT, 16, CC);
    kfinal<<<EBLK, TPB, 0, stream>>>(kbuf, abuf, wlog, mapre, mkpre, ma2, mk2, ma0, mk0);

    // --- sequential RWKV-7 scan ---
    wkv7_scan<<<BB * HH, 64, 0, stream>>>(rbuf, wlog, kbuf, vbuf, kkbuf, abuf, ybuf);

    // --- groupnorm + bonus + gate ---
    gn_bonus_gate<<<MTOT * HH, 64, 0, stream>>>(ybuf, rbuf, kbuf, vbuf, gbuf,
                                                ln_s, ln_b, faaaa, yg16);

    // --- output projection ---
    gemm(yg16, Wo16, out, MTOT, CC, CC);
}
